// CustomRNN_42451456754080
// MI455X (gfx1250) — matrix operand order repaired and checked by reading, not compiled, not run on hardware
//
#include <hip/hip_runtime.h>
#include <cstdint>
#include <cstddef>

// ---------------- types ----------------
typedef __attribute__((ext_vector_type(16))) __bf16 v16bf;
typedef __attribute__((ext_vector_type(8)))  __bf16 v8bf;
typedef __attribute__((ext_vector_type(8)))  float  v8f;

union BF16x16 { v16bf v; v8bf h[2]; };

// ---------------- dims ----------------
#define SEQ_T 2048
#define IN_I  2048
#define HID   4096
#define NG    16384   /* 4*HID */
#define NWG_REC 256   /* 256 WGs x 4 waves: spread L2 pull across WGPs */

// ---------------- helpers ----------------
__device__ __forceinline__ __bf16 f2bf(float f) {
  unsigned u = __builtin_bit_cast(unsigned, f);
  unsigned r = u + 0x7FFFu + ((u >> 16) & 1u);   // round-to-nearest-even
  return __builtin_bit_cast(__bf16, (unsigned short)(r >> 16));
}
__device__ __forceinline__ float bf2f(__bf16 b) {
  unsigned u = ((unsigned)__builtin_bit_cast(unsigned short, b)) << 16;
  return __builtin_bit_cast(float, u);
}
__device__ __forceinline__ float sigf(float x) { return 1.0f / (1.0f + __expf(-x)); }

// CDNA5 async global->LDS copy (ASYNCcnt path), 16B per lane, no VGPR staging.
// LDS offset = low 32 bits of the generic pointer (LDS aperture is in the high dword).
__device__ __forceinline__ void async_g2l_b128(const __bf16* gsrc, __bf16* lds_dst) {
  uint32_t ldsoff = (uint32_t)(uintptr_t)lds_dst;
  asm volatile("global_load_async_to_lds_b128 %0, %1, off"
               :: "v"(ldsoff), "v"(gsrc) : "memory");
}
#define WAIT_ASYNCCNT(n) asm volatile("s_wait_asynccnt " #n ::: "memory")

// ---------------- fp32 -> bf16 conversion ----------------
__global__ void cvt_bf16(const float* __restrict__ in, __bf16* __restrict__ out, size_t n) {
  size_t i = (size_t)blockIdx.x * blockDim.x + threadIdx.x;
  size_t stride = (size_t)gridDim.x * blockDim.x;
  for (; i < n; i += stride) out[i] = f2bf(in[i]);
}

// ---------------- init: zero h double-buffer + barrier counter ----------------
__global__ void init_state(__bf16* hbuf, unsigned* cnt) {
  int i = blockIdx.x * blockDim.x + threadIdx.x;
  if (i < 2 * HID) hbuf[i] = __builtin_bit_cast(__bf16, (unsigned short)0);
  if (i == 0) *cnt = 0u;
}

// ---------------- Phase 1: Gx[t,n] = sum_k X[t,k]*W_ih[n,k] + b_ih[n] + b_hh[n] ----------------
// Block tile 128(M) x 128(N), K-step 32, 8 waves in 2(M)x4(N) grid, wave tile 64x32.
// Triple-buffered LDS fed by async copies -> one barrier per K-step.
#define LDA 40  // padded LDS row stride in bf16 elems (32 + 8)
#define NKI (IN_I / 32)

__global__ __launch_bounds__(256) void gemm_gx(
    const __bf16* __restrict__ Xb, const __bf16* __restrict__ Wb,
    const float* __restrict__ b_ih, const float* __restrict__ b_hh,
    float* __restrict__ Gx) {
  __shared__ __bf16 As[3][128 * LDA];
  __shared__ __bf16 Bs[3][128 * LDA];

  const int tid  = threadIdx.x;
  const int lane = tid & 31;
  const int wave = tid >> 5;
  const int half = lane >> 4;
  const int l16  = lane & 15;
  const int mblk = blockIdx.y;   // 0..15
  const int nblk = blockIdx.x;   // 0..127
  const int wm   = wave & 1;     // 0..1
  const int wn   = wave >> 1;    // 0..3

  // per-thread 16B chunk coords (2 chunks each for A and B tiles; 512 chunks/tile)
  const int row0 = tid >> 2;
  const int col0 = (tid & 3) * 8;
  const int row1 = 64 + (tid >> 2);
  const int col1 = col0;

  const __bf16* gA = Xb + (size_t)(mblk * 128) * IN_I;
  const __bf16* gB = Wb + (size_t)(nblk * 128) * IN_I;

  auto issue_tile = [&](int kidx, int buf) {
    const int kb = kidx * 32;
    async_g2l_b128(&gA[(size_t)row0 * IN_I + kb + col0], &As[buf][row0 * LDA + col0]);
    async_g2l_b128(&gA[(size_t)row1 * IN_I + kb + col1], &As[buf][row1 * LDA + col1]);
    async_g2l_b128(&gB[(size_t)row0 * IN_I + kb + col0], &Bs[buf][row0 * LDA + col0]);
    async_g2l_b128(&gB[(size_t)row1 * IN_I + kb + col1], &Bs[buf][row1 * LDA + col1]);
  };

  v8f acc[4][2];
#pragma unroll
  for (int a = 0; a < 4; ++a)
#pragma unroll
    for (int b = 0; b < 2; ++b) {
      v8f z = {0.f,0.f,0.f,0.f,0.f,0.f,0.f,0.f};
      acc[a][b] = z;
    }

  issue_tile(0, 0);

  for (int ki = 0; ki < NKI; ++ki) {
    const int buf = ki % 3;
    if (ki + 1 < NKI) {
      issue_tile(ki + 1, (ki + 1) % 3);
      WAIT_ASYNCCNT(4);   // my 4 copies for tile ki are done
    } else {
      WAIT_ASYNCCNT(0);
    }
    __syncthreads();      // everyone's tile-ki copies landed; prior readers long done

    BF16x16 af[4], bfr[2];
#pragma unroll
    for (int mf = 0; mf < 4; ++mf) {
      int row = wm * 64 + mf * 16 + l16;
      af[mf].h[0] = *(const v8bf*)&As[buf][row * LDA + half * 8];
      af[mf].h[1] = *(const v8bf*)&As[buf][row * LDA + 16 + half * 8];
    }
#pragma unroll
    for (int nf = 0; nf < 2; ++nf) {
      int row = wn * 32 + nf * 16 + l16;
      bfr[nf].h[0] = *(const v8bf*)&Bs[buf][row * LDA + half * 8];
      bfr[nf].h[1] = *(const v8bf*)&Bs[buf][row * LDA + 16 + half * 8];
    }
#pragma unroll
    for (int mf = 0; mf < 4; ++mf)
#pragma unroll
      for (int nf = 0; nf < 2; ++nf)
        acc[mf][nf] = __builtin_amdgcn_wmma_f32_16x16x32_bf16(
            false, af[mf].v, false, bfr[nf].v, (short)0, acc[mf][nf], false, false);
  }

  // epilogue: add fused biases, stream out (non-temporal so Whh stays in L2 later)
#pragma unroll
  for (int mf = 0; mf < 4; ++mf)
#pragma unroll
    for (int nf = 0; nf < 2; ++nf) {
      int n = nblk * 128 + wn * 32 + nf * 16 + l16;
      float bias = b_ih[n] + b_hh[n];
#pragma unroll
      for (int r = 0; r < 8; ++r) {
        int m = mblk * 128 + wm * 64 + mf * 16 + r + 8 * half;
        __builtin_nontemporal_store(acc[mf][nf][r] + bias, &Gx[(size_t)m * NG + n]);
      }
    }
}

// ---------------- Phase 2: persistent LSTM recurrence ----------------
// 256 WGs x 128 thr (4 waves). WG w owns h[w*16 .. w*16+16).
// Wave q in {i,f,g,o} -> 16 gate rows q*HID + w*16 + m.
// B-matrix trick: broadcast h chunk to all 16 columns -> all C columns identical.
__global__ __launch_bounds__(128) void lstm_rec(
    const __bf16* __restrict__ Whhb, const float* __restrict__ Gx,
    const __bf16* __restrict__ Wlinb, const float* __restrict__ blin,
    __bf16* hbuf /* 2*HID, double buffered */, unsigned* cnt,
    float* __restrict__ out) {
  const int tid  = threadIdx.x;
  const int lane = tid & 31;
  const int q    = tid >> 5;           // wave = gate quadrant 0..3
  const int half = lane >> 4;
  const int l16  = lane & 15;
  const int w    = blockIdx.x;         // 0..255
  const int rowbase = q * HID + w * 16;

  __shared__ float gl[4][16];
  __shared__ float red[128];

  float c = 0.f;  // valid for tid<16: cell state c[w*16+tid]

  const __bf16* wrow = Whhb + (size_t)(rowbase + l16) * HID;

  for (int t = 0; t < SEQ_T; ++t) {
    const __bf16* hr = hbuf + (size_t)(t & 1) * HID;     // h_{t-1}

    // accumulators seeded from Gx (biases already fused); dual acc hides WMMA latency
    v8f acc0, acc1 = {0.f,0.f,0.f,0.f,0.f,0.f,0.f,0.f};
    const float* gx = Gx + (size_t)t * NG + rowbase;
#pragma unroll
    for (int r = 0; r < 8; ++r)
      acc0[r] = __builtin_nontemporal_load(gx + r + 8 * half);

    for (int kk = 0; kk < HID; kk += 64) {
      BF16x16 a0, a1, b0, b1;
      a0.h[0] = *(const v8bf*)(wrow + kk + half * 8);
      a0.h[1] = *(const v8bf*)(wrow + kk + 16 + half * 8);
      a1.h[0] = *(const v8bf*)(wrow + kk + 32 + half * 8);
      a1.h[1] = *(const v8bf*)(wrow + kk + 48 + half * 8);
      b0.h[0] = *(const v8bf*)(hr + kk + half * 8);
      b0.h[1] = *(const v8bf*)(hr + kk + 16 + half * 8);
      b1.h[0] = *(const v8bf*)(hr + kk + 32 + half * 8);
      b1.h[1] = *(const v8bf*)(hr + kk + 48 + half * 8);
      acc0 = __builtin_amdgcn_wmma_f32_16x16x32_bf16(false, a0.v, false, b0.v,
                                                     (short)0, acc0, false, false);
      acc1 = __builtin_amdgcn_wmma_f32_16x16x32_bf16(false, a1.v, false, b1.v,
                                                     (short)0, acc1, false, false);
    }
#pragma unroll
    for (int r = 0; r < 8; ++r) acc0[r] += acc1[r];

    // lanes 0 and 16 hold column 0 (== every column); extract 16 gate values
    if (l16 == 0) {
#pragma unroll
      for (int r = 0; r < 8; ++r) gl[q][r + 8 * half] = acc0[r];
    }
    __syncthreads();

    // local nonlinearity + state update for our 16-element slice
    if (tid < 16) {
      float iv = sigf(gl[0][tid]);
      float fv = sigf(gl[1][tid]);
      float gv = tanhf(gl[2][tid]);
      float ov = sigf(gl[3][tid]);
      c = fv * c + iv * gv;
      float hval = ov * tanhf(c);
      hbuf[(size_t)((t + 1) & 1) * HID + w * 16 + tid] = f2bf(hval);
    }

    // one grid-wide barrier per step (sense via cumulative counter)
    __threadfence();
    __syncthreads();
    if (tid == 0) {
      atomicAdd(cnt, 1u);
      unsigned target = (unsigned)(t + 1) * NWG_REC;
      while (atomicAdd(cnt, 0u) < target) __builtin_amdgcn_s_sleep(1);
    }
    __syncthreads();
    __threadfence();

    // WG0: scalar readout p[t] = sigmoid(W_lin . h_t + b_lin)
    if (w == 0) {
      const __bf16* hn = hbuf + (size_t)((t + 1) & 1) * HID;
      float part = 0.f;
#pragma unroll
      for (int u = 0; u < 32; ++u) {
        int k = tid * 32 + u;
        part += bf2f(Wlinb[k]) * bf2f(hn[k]);
      }
      red[tid] = part;
      __syncthreads();
      for (int off = 64; off > 0; off >>= 1) {
        if (tid < off) red[tid] += red[tid + off];
        __syncthreads();
      }
      if (tid == 0) out[t] = sigf(red[0] + blin[0]);
    }
  }
}

// ---------------- host launcher ----------------
extern "C" void kernel_launch(void* const* d_in, const int* in_sizes, int n_in,
                              void* d_out, int out_size, void* d_ws, size_t ws_size,
                              hipStream_t stream) {
  const float* X    = (const float*)d_in[0];  // [T, I]
  const float* Wih  = (const float*)d_in[1];  // [4H, I]
  const float* Whh  = (const float*)d_in[2];  // [4H, H]
  const float* bih  = (const float*)d_in[3];  // [4H]
  const float* bhh  = (const float*)d_in[4];  // [4H]
  const float* Wlin = (const float*)d_in[5];  // [1, H]
  const float* blin = (const float*)d_in[6];  // [1]
  float* out = (float*)d_out;

  uintptr_t p = (uintptr_t)d_ws;
  auto carve = [&](size_t bytes) -> void* {
    p = (p + 255) & ~(uintptr_t)255;
    void* r = (void*)p;
    p += bytes;
    return r;
  };
  __bf16*  Xb    = (__bf16*)carve((size_t)SEQ_T * IN_I * 2);
  __bf16*  Wihb  = (__bf16*)carve((size_t)NG * IN_I * 2);
  __bf16*  Whhb  = (__bf16*)carve((size_t)NG * HID * 2);   // 128 MB: L2-resident
  __bf16*  Wlinb = (__bf16*)carve((size_t)HID * 2);
  __bf16*  hbuf  = (__bf16*)carve((size_t)2 * HID * 2);
  float*   Gx    = (float*)carve((size_t)SEQ_T * NG * 4);
  unsigned* cnt  = (unsigned*)carve(256);

  cvt_bf16<<<2048, 256, 0, stream>>>(X,    Xb,    (size_t)SEQ_T * IN_I);
  cvt_bf16<<<4096, 256, 0, stream>>>(Wih,  Wihb,  (size_t)NG * IN_I);
  cvt_bf16<<<8192, 256, 0, stream>>>(Whh,  Whhb,  (size_t)NG * HID);
  cvt_bf16<<<16,   256, 0, stream>>>(Wlin, Wlinb, (size_t)HID);
  init_state<<<32, 256, 0, stream>>>(hbuf, cnt);

  dim3 gg(NG / 128, SEQ_T / 128);  // 128 x 16 workgroups
  gemm_gx<<<gg, 256, 0, stream>>>(Xb, Wihb, bih, bhh, Gx);

  lstm_rec<<<NWG_REC, 128, 0, stream>>>(Whhb, Gx, Wlinb, blin, hbuf, cnt, out);
}
